// Differentiable_Xinanjiang_23742579212559
// MI455X (gfx1250) — matrix operand order repaired and checked by reading, not compiled, not run on hardware
//
#include <hip/hip_runtime.h>

// ---------------------------------------------------------------------------
// Differentiable Xinanjiang on MI455X (gfx1250, wave32, WMMA).
//
// Phase 1 (WMMA): params = sigmoid(tanh(tanh(attrs@w1+b1)@w2+b2)@w3+b3)
//   524288 rows. Two 16-row tiles (M=32) per iteration so each LDS-resident
//   weight B-fragment feeds two v_wmma_f32_16x16x32_f16 and the dual
//   accumulator chains fill the WMMA->VALU hazard slots. Elementwise tanh
//   uses gfx1250's hardware v_tanh_f32 (TRANS pipe, co-executes with WMMA).
// Phase 2 (serial): per-batch scan over T=2048 (nonlinear recurrence),
//   independent of phase 1 -> fused via block split so it overlaps the GEMM.
//   heaviside == sigmoid(20x) == fma(.5, tanh(10x), .5): 1 TRANS per h.
// Phase 3: elementwise runoff/partition -> total_q.
// ---------------------------------------------------------------------------

#define Bdim   256
#define Tdim   2048
#define Fdim   20
#define NROWS  (Bdim * Tdim)      // 524288
#define NTILES (NROWS / 16)       // 32768
#define NPAIRS (NTILES / 2)       // 16384  (M=32 per iteration)

#define MLP_BLOCKS  1024          // 1 wave each -> 16 tile-pairs per block
#define SCAN_BLOCKS 8             // 8 x 32 lanes = 256 chains

// LDS strides (f16 elements); multiples of 8 so every 16B fragment read is
// aligned; 264/72 also give conflict-free half-wave bank patterns.
#define W1S 16     // w1 col-major [256][16], k 0..14 valid, k=15 zero
#define W2S 264    // w2 col-major [64][264]
#define W3S 72     // w3 col-major [16][72], cols 8..15 zero
#define A1S 264    // h1 activations f16 [32][264]
#define A2S 72     // h2 activations f16 [32][72] (aliases h1 buffer)

typedef __attribute__((ext_vector_type(16))) _Float16 v16h;
typedef __attribute__((ext_vector_type(8)))  float    v8f;

__device__ __forceinline__ float fexp(float x) { return __builtin_amdgcn_exp2f(x * 1.44269504088896341f); }
__device__ __forceinline__ float frcp(float x) { return __builtin_amdgcn_rcpf(x); }

#if __has_builtin(__builtin_amdgcn_tanhf)
// gfx1250 hardware transcendental: single v_tanh_f32.
__device__ __forceinline__ float ftanh(float x) { return __builtin_amdgcn_tanhf(x); }
#else
__device__ __forceinline__ float ftanh(float x) {
  float xc = fminf(fmaxf(x, -30.f), 30.f);
  float t  = fexp(2.f * xc);
  return (t - 1.f) * frcp(t + 1.f);
}
#endif
// sigmoid(x) = 0.5*tanh(x/2) + 0.5
__device__ __forceinline__ float fsigmoid(float x) { return fmaf(0.5f, ftanh(0.5f * x), 0.5f); }
// reference heaviside: (tanh(10x)+1)/2
__device__ __forceinline__ float heav(float x) { return fmaf(0.5f, ftanh(10.f * x), 0.5f); }
__device__ __forceinline__ float clip1e5(float x) { return fminf(fmaxf(x, -100000.f), 100000.f); }

__device__ __forceinline__ v8f wmma_f16(v16h a, v16h b, v8f c) {
  return __builtin_amdgcn_wmma_f32_16x16x32_f16(false, a, false, b, (short)0, c, false, false);
}
// B 32x16 f16 fragment: lane = column N, element e -> K = 16*half + e
__device__ __forceinline__ v16h load_bfrag(const _Float16* wp) {
  v16h b;
  #pragma unroll
  for (int e = 0; e < 16; ++e) b[e] = wp[e];
  return b;
}
// A 16x32 f16 fragment from row-major LDS: rp = row base + chunk + 16*half
// element e -> K=16h+e (rp[e])
__device__ __forceinline__ v16h load_afrag(const _Float16* rp) {
  v16h a;
  #pragma unroll
  for (int e = 0; e < 16; ++e) a[e] = rp[e];
  return a;
}

__global__ __launch_bounds__(32)
void xaj_mlp_scan_kernel(const float* __restrict__ in,
                         const float* __restrict__ w1, const float* __restrict__ b1,
                         const float* __restrict__ w2, const float* __restrict__ b2,
                         const float* __restrict__ w3, const float* __restrict__ b3,
                         float* __restrict__ params,
                         float* __restrict__ wu_o, float* __restrict__ wl_o,
                         float* __restrict__ wd_o)
{
  __shared__ _Float16 w1c[256 * W1S];   //  8192 B
  __shared__ _Float16 w2c[64 * W2S];    // 33792 B
  __shared__ _Float16 w3c[16 * W3S];    //  2304 B
  __shared__ float    b1s[256];
  __shared__ float    b2s[64];
  __shared__ float    b3s[16];
  __shared__ _Float16 h1s[32 * A1S];    // 16896 B; reused for h2 [32][A2S]
                                        // total ~62.5 KB static LDS

  const int lane = threadIdx.x;         // blockDim.x == 32 (one wave)

  if (blockIdx.x < MLP_BLOCKS) {
    // ---------------- weight preprocessing: f32 global -> f16 col-major LDS
    for (int i = lane; i < 256 * W1S; i += 32) w1c[i] = (_Float16)0.f;
    for (int i = lane; i < 64 * W2S;  i += 32) w2c[i] = (_Float16)0.f;
    for (int i = lane; i < 16 * W3S;  i += 32) w3c[i] = (_Float16)0.f;
    for (int i = lane; i < 15 * 256; i += 32) { int k = i / 256, n = i % 256; w1c[n * W1S + k] = (_Float16)w1[i]; }
    for (int i = lane; i < 256 * 64; i += 32) { int k = i / 64,  n = i % 64;  w2c[n * W2S + k] = (_Float16)w2[i]; }
    for (int i = lane; i < 64 * 8;   i += 32) { int k = i / 8,   n = i % 8;   w3c[n * W3S + k] = (_Float16)w3[i]; }
    for (int i = lane; i < 256; i += 32) b1s[i] = b1[i];
    for (int i = lane; i < 64;  i += 32) b2s[i] = b2[i];
    for (int i = lane; i < 16;  i += 32) b3s[i] = (i < 8) ? b3[i] : 0.f;
    __syncthreads();   // single-wave WG -> S_NOP; same-wave LDS is in-order

    const int half = lane >> 4;         // 0: lanes 0..15, 1: lanes 16..31
    const int l16  = lane & 15;

    for (int pair = blockIdx.x; pair < NPAIRS; pair += MLP_BLOCKS) {
      const int rowbase = pair * 32;    // two 16-row tiles

      // ---- layer-1 A fragments straight from global (features 5..19)
      v16h a1[2];
      #pragma unroll
      for (int t = 0; t < 2; ++t) {
        const float* arow = in + (size_t)(rowbase + 16 * t + l16) * Fdim + 5;
        #pragma unroll
        for (int e = 0; e < 16; ++e) {
          int k = 16 * half + e;
          float v = (k < 15) ? arow[k] : 0.f;
          a1[t][e] = (_Float16)v;
        }
      }

      // ---- layer 1: h1 = tanh(A @ w1 + b1), N = 256 -> 16 x 2 WMMAs
      #pragma unroll
      for (int nt = 0; nt < 16; ++nt) {
        const int N = nt * 16 + l16;
        const float bias = b1s[N];
        // K = 16*half + e; real K stops at 15 -> upper half-wave is zero pad.
        v16h b;
        if (half == 0) {
          b = load_bfrag(&w1c[N * W1S]);
        } else {
          #pragma unroll
          for (int e = 0; e < 16; ++e) b[e] = (_Float16)0.f;
        }
        v8f c0, c1;
        #pragma unroll
        for (int j = 0; j < 8; ++j) { c0[j] = bias; c1[j] = bias; }
        c0 = wmma_f16(a1[0], b, c0);
        c1 = wmma_f16(a1[1], b, c1);
        #pragma unroll
        for (int j = 0; j < 8; ++j) {
          h1s[(j + 8 * half) * A1S + N]      = (_Float16)ftanh(c0[j]);
          h1s[(j + 8 * half + 16) * A1S + N] = (_Float16)ftanh(c1[j]);
        }
      }

      // ---- layer-2 A fragments (2 tiles x 8 K-chunks) from h1 LDS.
      // All register-resident before h2 overwrites the h1 buffer below.
      v16h a2[2][8];
      #pragma unroll
      for (int t = 0; t < 2; ++t)
        #pragma unroll
        for (int c2 = 0; c2 < 8; ++c2)
          a2[t][c2] = load_afrag(&h1s[(l16 + 16 * t) * A1S + 32 * c2 + 16 * half]);

      // ---- layer 2: h2 = tanh(h1 @ w2 + b2), N = 64, K = 256 -> 4 x 8 x 2
      #pragma unroll
      for (int nt = 0; nt < 4; ++nt) {
        const int N = nt * 16 + l16;
        const float bias = b2s[N];
        v8f c0, c1;
        #pragma unroll
        for (int j = 0; j < 8; ++j) { c0[j] = bias; c1[j] = bias; }
        #pragma unroll
        for (int c2 = 0; c2 < 8; ++c2) {
          v16h b = load_bfrag(&w2c[N * W2S + 32 * c2 + 16 * half]);
          c0 = wmma_f16(a2[0][c2], b, c0);
          c1 = wmma_f16(a2[1][c2], b, c1);
        }
        #pragma unroll
        for (int j = 0; j < 8; ++j) {   // h2 reuses the h1 buffer (stride A2S)
          h1s[(j + 8 * half) * A2S + N]      = (_Float16)ftanh(c0[j]);
          h1s[(j + 8 * half + 16) * A2S + N] = (_Float16)ftanh(c1[j]);
        }
      }

      // ---- layer 3: params = sigmoid(h2 @ w3 + b3), K = 64, N = 8 (pad 16)
      v16h a3[2][2];
      #pragma unroll
      for (int t = 0; t < 2; ++t)
        #pragma unroll
        for (int c3 = 0; c3 < 2; ++c3)
          a3[t][c3] = load_afrag(&h1s[(l16 + 16 * t) * A2S + 32 * c3 + 16 * half]);
      {
        const int N = l16;
        const float bias = b3s[N];
        v8f c0, c1;
        #pragma unroll
        for (int j = 0; j < 8; ++j) { c0[j] = bias; c1[j] = bias; }
        #pragma unroll
        for (int c3 = 0; c3 < 2; ++c3) {
          v16h b = load_bfrag(&w3c[N * W3S + 32 * c3 + 16 * half]);
          c0 = wmma_f16(a3[0][c3], b, c0);
          c1 = wmma_f16(a3[1][c3], b, c1);
        }
        if (l16 < 8) {
          #pragma unroll
          for (int j = 0; j < 8; ++j) {
            params[(size_t)(rowbase + j + 8 * half) * 8 + l16]      = fsigmoid(c0[j]);
            params[(size_t)(rowbase + 16 + j + 8 * half) * 8 + l16] = fsigmoid(c1[j]);
          }
        }
      }
    }
  } else {
    // ---------------- sequential scan: 256 independent chains, one per lane
    const int bidx = (blockIdx.x - MLP_BLOCKS) * 32 + lane;   // 0..255
    const float* base = in + (size_t)bidx * Tdim * Fdim;
    float wu = 0.f, wl = 0.f, wd = 0.f;
    float pet = base[0];
    float p   = base[2];
    for (int t = 0; t < Tdim; ++t) {
      // software pipeline: fetch next step's forcings before the dep chain
      float pet_n = 0.f, p_n = 0.f;
      if (t + 1 < Tdim) {
        pet_n = base[(size_t)(t + 1) * Fdim];
        p_n   = base[(size_t)(t + 1) * Fdim + 2];
      }
      if (t + 16 < Tdim)
        __builtin_prefetch(&base[(size_t)(t + 16) * Fdim], 0, 0);

      float et1  = heav(wu - pet) * pet + heav(pet - wu) * wu;
      float r1   = pet - et1;
      float rp   = heav(r1) * r1;
      float et22 = heav(rp - wl) * wl + heav(wl - rp) * rp;
      float et2  = heav(rp) * et22;
      float r2   = rp - et2;
      float et33 = heav(r2 - wd) * wd + heav(wd - r2) * r2;
      float et3  = heav(r2) * et33;

      wu += clip1e5(p - et1);
      wl += clip1e5(p - et1 - et2);
      wd += clip1e5(p - et1 - et2 - et3);

      const size_t idx = (size_t)bidx * Tdim + t;
      wu_o[idx] = wu; wl_o[idx] = wl; wd_o[idx] = wd;
      pet = pet_n; p = p_n;
    }
  }
}

__global__ void xaj_final_kernel(const float* __restrict__ in,
                                 const float* __restrict__ params,
                                 const float* __restrict__ wu_a,
                                 const float* __restrict__ wl_a,
                                 const float* __restrict__ wd_a,
                                 float* __restrict__ out)
{
  const size_t idx = (size_t)blockIdx.x * blockDim.x + threadIdx.x;
  if (idx >= (size_t)NROWS) return;
  const float p = in[idx * Fdim + 2];
  const float* pr = params + idx * 8;
  const float wum = pr[0], wlm = pr[1], wdm = pr[2];
  const float bb = pr[3], cc = pr[4];
  const float k1 = pr[5], k2 = pr[6], k3 = pr[7];
  const float wu = wu_a[idx], wl = wl_a[idx], wd = wd_a[idx];

  // reference calls _runoff(wu, wd, wl, p, wum, wdm, wlm, b, c):
  // state wl/wd swapped and params wlm/wdm swapped at the call site.
  const float WU = wu, WL = wd;               // function-local wu, wl
  const float WUM = wum * 19.9f + 0.1f;
  const float WLM = wdm * 30.f + 60.f;        // wlm arg <- wdm param
  const float WDM = wlm * 60.f + 60.f;        // wdm arg <- wlm param
  const float C  = cc * 0.19f + 0.01f;
  const float Bc = bb * 0.3f  + 0.1f;
  const float iw = frcp(WUM + WLM + WDM);
  const float t0 = WU * iw, t1 = WL * iw;
  const float s  = C * t0 * t0 + Bc * t1 * t1;
  const float ps = p - s;
  const float runoff = heav(ps) * ps;

  const float K1 = k1 * 0.69f + 0.01f;
  const float K2 = k2 * 0.69f + 0.01f;
  const float K3 = k3 * 0.89f + 0.01f;
  const float sr  = K1 * runoff;
  const float itf = K2 * (runoff - sr);
  const float bf  = K3 * (runoff - sr - itf);
  out[idx] = sr + 0.5f * itf + 0.25f * bf;
}

extern "C" void kernel_launch(void* const* d_in, const int* in_sizes, int n_in,
                              void* d_out, int out_size, void* d_ws, size_t ws_size,
                              hipStream_t stream) {
  const float* in = (const float*)d_in[0];
  const float* w1 = (const float*)d_in[1];
  const float* b1 = (const float*)d_in[2];
  const float* w2 = (const float*)d_in[3];
  const float* b2 = (const float*)d_in[4];
  const float* w3 = (const float*)d_in[5];
  const float* b3 = (const float*)d_in[6];
  float* out = (float*)d_out;

  // workspace layout: params (16 MB) | wu | wl | wd (2 MB each) = 22 MB
  char* ws = (char*)d_ws;
  float* params = (float*)ws;
  float* wu = (float*)(ws + (size_t)NROWS * 8 * sizeof(float));
  float* wl = wu + NROWS;
  float* wd = wl + NROWS;

  // MLP (WMMA) and the serial scan are independent: run them as one grid so
  // the 0.1ms-class serial recurrence overlaps the GEMM blocks.
  xaj_mlp_scan_kernel<<<MLP_BLOCKS + SCAN_BLOCKS, 32, 0, stream>>>(
      in, w1, b1, w2, b2, w3, b3, params, wu, wl, wd);
  xaj_final_kernel<<<(NROWS + 255) / 256, 256, 0, stream>>>(
      in, params, wu, wl, wd, out);
}
